// DomainAdaptationLayer_45492293599520
// MI455X (gfx1250) — compile-verified, matrix operand order checked by reading, not run on hardware
//
#include <hip/hip_runtime.h>
#include <cstdint>
#include <cstddef>

// ---------------------------------------------------------------------------
// DomainAdaptationLayer fused kernel for gfx1250 (MI455X).
//   out[0 .. B*512)          = subject LayerNorm(x)         (f32, exact)
//   out[B*512 .. B*512+B*16) = MLP domain logits            (f32, hi/lo f16 WMMA)
// ---------------------------------------------------------------------------

#define D_IN  512
#define H1D   256
#define H2D   128
#define NSUB  16
#define TILE_ROWS 64          // rows per block (4 waves x 16 rows)
#define H1_STRIDE (H1D + 6)   // dword stride, even (b64-aligned) + bank skew
#define H2_STRIDE (H2D + 6)

typedef _Float16 v16h __attribute__((ext_vector_type(16)));
typedef float    v8f  __attribute__((ext_vector_type(8)));

__device__ __forceinline__ v8f wmma_f16(v16h a, v16h b, v8f c) {
  // D(16x16 f32) = A(16x32 f16) * B(32x16 f16) + C
  return __builtin_amdgcn_wmma_f32_16x16x32_f16(false, a, false, b, (short)0, c,
                                                false, false);
}

__device__ __forceinline__ float gelu_exact(float v) {
  return 0.5f * v * (1.0f + erff(v * 0.7071067811865475f));
}

// ---------------------------------------------------------------------------
// Pre-swizzle a row-major f32 weight W[K][N] into WMMA B-fragment order,
// split into f16 hi + f16 lo halves (W = hi + lo to ~22 mantissa bits).
// Slot layout (u32 units): ((ntile*KS + kstep)*32 + lane)*8 + v
//   lane -> col = ntile*16 + (lane&15), K window base = (lane>>4)*8
//   v    -> k   = kstep*32 + (v<4 ? 0 : 16) + kbase + 2*(v&3)   (pair k,k+1)
// So the GEMM reads one contiguous 32B v16h per lane per fragment.
// ---------------------------------------------------------------------------
__global__ void prep_wfrag(const float* __restrict__ W, int K, int N,
                           uint32_t* __restrict__ outHi,
                           uint32_t* __restrict__ outLo) {
  int t = blockIdx.x * blockDim.x + threadIdx.x;
  int KS = K >> 5;
  int total = (N >> 4) * KS * 32 * 8;
  if (t >= total) return;
  int v    = t & 7;
  int lane = (t >> 3) & 31;
  int fs   = t >> 8;            // ntile*KS + kstep
  int ks   = fs % KS;
  int nt   = fs / KS;
  int k   = (ks << 5) + ((v < 4) ? 0 : 16) + ((lane >> 4) << 3) + ((v & 3) << 1);
  int col = (nt << 4) + (lane & 15);
  float f0 = W[(size_t)k * N + col];
  float f1 = W[(size_t)(k + 1) * N + col];
  _Float16 h0 = (_Float16)f0, h1 = (_Float16)f1;
  _Float16 l0 = (_Float16)(f0 - (float)h0);
  _Float16 l1 = (_Float16)(f1 - (float)h1);
  union { _Float16 h[2]; uint32_t u; } hu, lu;
  hu.h[0] = h0; hu.h[1] = h1;
  lu.h[0] = l0; lu.h[1] = l1;
  outHi[t] = hu.u;
  outLo[t] = lu.u;
}

// ---------------------------------------------------------------------------
// Fused LayerNorm + 3-layer MLP. 128 threads = 4 waves; wave w owns rows
// [tile + 16w, tile + 16w + 16). All LDS regions are wave-private slices, so
// no block barriers are needed (in-wave DS ordering is enough).
// ---------------------------------------------------------------------------
__global__ __launch_bounds__(128, 1)
void fused_ln_mlp(const float* __restrict__ x,
                  const int*   __restrict__ groups,
                  const float* __restrict__ gammas,
                  const float* __restrict__ betas,
                  const float* __restrict__ b1,
                  const float* __restrict__ b2,
                  const float* __restrict__ b3,
                  const uint32_t* __restrict__ w1hi, const uint32_t* __restrict__ w1lo,
                  const uint32_t* __restrict__ w2hi, const uint32_t* __restrict__ w2lo,
                  const uint32_t* __restrict__ w3hi, const uint32_t* __restrict__ w3lo,
                  float* __restrict__ out_norm,
                  float* __restrict__ out_logits) {
  __shared__ float h1s[TILE_ROWS * H1_STRIDE];   // ~67 KB
  __shared__ float h2s[TILE_ROWS * H2_STRIDE];   // ~34 KB

  const int lane  = threadIdx.x & 31;
  const int w     = threadIdx.x >> 5;
  const int tile0 = blockIdx.x * TILE_ROWS;
  const int wrow0 = tile0 + (w << 4);   // first global row of this wave
  const int lrow0 = (w << 4);           // first LDS row of this wave
  const int rA    = lane & 15;          // WMMA A/B row-col lane id
  const int kb    = (lane >> 4) << 3;   // K window base within 32-wide step
  const int hiH   = lane >> 4;          // C/D half select

  // ---- Phase 1: exact f32 LayerNorm, one wave per row, coalesced ----------
  for (int r = 0; r < 16; ++r) {
    const int row = wrow0 + r;
    const float* xr = x + (size_t)row * D_IN;
    float4 xv[4];
    float s = 0.f, sq = 0.f;
#pragma unroll
    for (int p = 0; p < 4; ++p) {
      xv[p] = *(const float4*)(xr + p * 128 + lane * 4);
      s  += xv[p].x + xv[p].y + xv[p].z + xv[p].w;
      sq += xv[p].x * xv[p].x + xv[p].y * xv[p].y +
            xv[p].z * xv[p].z + xv[p].w * xv[p].w;
    }
#pragma unroll
    for (int off = 16; off > 0; off >>= 1) {
      s  += __shfl_xor(s,  off, 32);
      sq += __shfl_xor(sq, off, 32);
    }
    const float mu  = s * (1.0f / 512.0f);
    const float var = sq * (1.0f / 512.0f) - mu * mu;
    const float rs  = rsqrtf(var + 1e-5f);
    const int g   = groups[row];
    const int idx = (g >= 0 && g < NSUB) ? g : NSUB;
    const float* gr = gammas + (size_t)idx * D_IN;
    const float* br = betas  + (size_t)idx * D_IN;
    float* outr = out_norm + (size_t)row * D_IN;
#pragma unroll
    for (int p = 0; p < 4; ++p) {
      const int col = p * 128 + lane * 4;
      float4 gv = *(const float4*)(gr + col);
      float4 bv = *(const float4*)(br + col);
      float4 o;
      o.x = (xv[p].x - mu) * rs * gv.x + bv.x;
      o.y = (xv[p].y - mu) * rs * gv.y + bv.y;
      o.z = (xv[p].z - mu) * rs * gv.z + bv.z;
      o.w = (xv[p].w - mu) * rs * gv.w + bv.w;
      *(float4*)(outr + col) = o;
    }
  }

  // ---- Phase 2: GEMM1  h1 = gelu(x @ W1 + b1)   [16 x 512] @ [512 x 256] --
  {
    v8f acc[16] = {};
    const float* xw = x + (size_t)(wrow0 + rA) * D_IN;
#pragma unroll 1
    for (int ks = 0; ks < 16; ++ks) {
      v16h ahi, alo;
#pragma unroll
      for (int v = 0; v < 8; ++v) {
        const int koff = (ks << 5) + ((v < 4) ? 0 : 16) + kb + ((v & 3) << 1);
        float2 f = *(const float2*)(xw + koff);
        _Float16 h0 = (_Float16)f.x, h1 = (_Float16)f.y;
        ahi[2 * v] = h0;                         ahi[2 * v + 1] = h1;
        alo[2 * v] = (_Float16)(f.x - (float)h0); alo[2 * v + 1] = (_Float16)(f.y - (float)h1);
      }
#pragma unroll
      for (int nt = 0; nt < 16; ++nt) {
        const size_t slot = ((size_t)((nt << 4) + ks) * 32 + lane) * 8;
        v16h bh = *(const v16h*)(w1hi + slot);
        v16h bl = *(const v16h*)(w1lo + slot);
        acc[nt] = wmma_f16(ahi, bh, acc[nt]);
        acc[nt] = wmma_f16(alo, bh, acc[nt]);
        acc[nt] = wmma_f16(ahi, bl, acc[nt]);
      }
    }
#pragma unroll
    for (int nt = 0; nt < 16; ++nt) {
      const int col = (nt << 4) + rA;
      const float bias = b1[col];
#pragma unroll
      for (int v = 0; v < 8; ++v) {
        float val = acc[nt][v] + bias;
        h1s[(size_t)(lrow0 + v + 8 * hiH) * H1_STRIDE + col] = gelu_exact(val);
      }
    }
  }

  // ---- Phase 3: GEMM2  h2 = gelu(h1 @ W2 + b2)  [16 x 256] @ [256 x 128] --
  {
    v8f acc[8] = {};
    const float* h1r = &h1s[(size_t)(lrow0 + rA) * H1_STRIDE];
#pragma unroll 1
    for (int ks = 0; ks < 8; ++ks) {
      v16h ahi, alo;
#pragma unroll
      for (int v = 0; v < 8; ++v) {
        const int koff = (ks << 5) + ((v < 4) ? 0 : 16) + kb + ((v & 3) << 1);
        float2 f = *(const float2*)(h1r + koff);
        _Float16 h0 = (_Float16)f.x, h1 = (_Float16)f.y;
        ahi[2 * v] = h0;                         ahi[2 * v + 1] = h1;
        alo[2 * v] = (_Float16)(f.x - (float)h0); alo[2 * v + 1] = (_Float16)(f.y - (float)h1);
      }
#pragma unroll
      for (int nt = 0; nt < 8; ++nt) {
        const size_t slot = ((size_t)(nt * 8 + ks) * 32 + lane) * 8;
        v16h bh = *(const v16h*)(w2hi + slot);
        v16h bl = *(const v16h*)(w2lo + slot);
        acc[nt] = wmma_f16(ahi, bh, acc[nt]);
        acc[nt] = wmma_f16(alo, bh, acc[nt]);
        acc[nt] = wmma_f16(ahi, bl, acc[nt]);
      }
    }
#pragma unroll
    for (int nt = 0; nt < 8; ++nt) {
      const int col = (nt << 4) + rA;
      const float bias = b2[col];
#pragma unroll
      for (int v = 0; v < 8; ++v) {
        float val = acc[nt][v] + bias;
        h2s[(size_t)(lrow0 + v + 8 * hiH) * H2_STRIDE + col] = gelu_exact(val);
      }
    }
  }

  // ---- Phase 4: GEMM3  logits = h2 @ W3 + b3    [16 x 128] @ [128 x 16] ---
  {
    v8f acc = {};
    const float* h2r = &h2s[(size_t)(lrow0 + rA) * H2_STRIDE];
#pragma unroll 1
    for (int ks = 0; ks < 4; ++ks) {
      v16h ahi, alo;
#pragma unroll
      for (int v = 0; v < 8; ++v) {
        const int koff = (ks << 5) + ((v < 4) ? 0 : 16) + kb + ((v & 3) << 1);
        float2 f = *(const float2*)(h2r + koff);
        _Float16 h0 = (_Float16)f.x, h1 = (_Float16)f.y;
        ahi[2 * v] = h0;                         ahi[2 * v + 1] = h1;
        alo[2 * v] = (_Float16)(f.x - (float)h0); alo[2 * v + 1] = (_Float16)(f.y - (float)h1);
      }
      const size_t slot = ((size_t)ks * 32 + lane) * 8;
      v16h bh = *(const v16h*)(w3hi + slot);
      v16h bl = *(const v16h*)(w3lo + slot);
      acc = wmma_f16(ahi, bh, acc);
      acc = wmma_f16(alo, bh, acc);
      acc = wmma_f16(ahi, bl, acc);
    }
    const float bias = b3[rA];
#pragma unroll
    for (int v = 0; v < 8; ++v) {
      out_logits[(size_t)(wrow0 + v + 8 * hiH) * 16 + rA] = acc[v] + bias;
    }
  }
}

// ---------------------------------------------------------------------------
extern "C" void kernel_launch(void* const* d_in, const int* in_sizes, int n_in,
                              void* d_out, int out_size, void* d_ws, size_t ws_size,
                              hipStream_t stream) {
  const float* x      = (const float*)d_in[0];
  const int*   groups = (const int*)  d_in[1];
  const float* gammas = (const float*)d_in[2];
  const float* betas  = (const float*)d_in[3];
  const float* W1     = (const float*)d_in[4];
  const float* b1     = (const float*)d_in[5];
  const float* W2     = (const float*)d_in[6];
  const float* b2     = (const float*)d_in[7];
  const float* W3     = (const float*)d_in[8];
  const float* b3     = (const float*)d_in[9];

  const int B = in_sizes[1];  // number of rows (131072)

  // Workspace layout (u32 units): swizzled hi/lo f16 weight fragments.
  uint32_t* ws   = (uint32_t*)d_ws;
  uint32_t* w1hi = ws;                 // 16*16*32*8 = 65536
  uint32_t* w1lo = ws + 65536;
  uint32_t* w2hi = ws + 131072;        //  8* 8*32*8 = 16384
  uint32_t* w2lo = ws + 147456;
  uint32_t* w3hi = ws + 163840;        //  1* 4*32*8 =  1024
  uint32_t* w3lo = ws + 164864;        // total 165888 u32 = 648 KB

  prep_wfrag<<<(65536 + 255) / 256, 256, 0, stream>>>(W1, 512, 256, w1hi, w1lo);
  prep_wfrag<<<(16384 + 255) / 256, 256, 0, stream>>>(W2, 256, 128, w2hi, w2lo);
  prep_wfrag<<<( 1024 + 255) / 256, 256, 0, stream>>>(W3, 128,  16, w3hi, w3lo);

  float* out_norm   = (float*)d_out;
  float* out_logits = out_norm + (size_t)B * D_IN;

  fused_ln_mlp<<<B / TILE_ROWS, 128, 0, stream>>>(
      x, groups, gammas, betas, b1, b2, b3,
      w1hi, w1lo, w2hi, w2lo, w3hi, w3lo, out_norm, out_logits);
}
